// MAB_49246095016375
// MI455X (gfx1250) — matrix operand order repaired and checked by reading, not compiled, not run on hardware
//
#include <hip/hip_runtime.h>
#include <hip/hip_bf16.h>

typedef __attribute__((ext_vector_type(16))) _Float16 v16h;
typedef __attribute__((ext_vector_type(8)))  _Float16 v8h;
typedef __attribute__((ext_vector_type(8)))  float    v8f;
typedef __attribute__((ext_vector_type(4)))  unsigned u32x4;
typedef __attribute__((ext_vector_type(8)))  int      i32x8;
typedef __attribute__((ext_vector_type(4)))  int      i32x4;

#define NB     2
#define NQS    2048
#define NKVS   2048
#define DM     1024
#define NH     16
#define DH     64
#define NKV1   (NKVS + 1)

// ---------- CDNA5 async memory->LDS (ASYNCcnt-tracked) ----------
// LDS operand = low 32 bits of the generic pointer (LDS aperture keeps the
// byte offset in addr[31:0], ISA 10.2).
__device__ __forceinline__ void async_b128(void* lds_ptr, const void* gaddr) {
    unsigned lo = (unsigned)(unsigned long long)lds_ptr;
    asm volatile("global_load_async_to_lds_b128 %0, %1, off"
                 :: "v"(lo), "v"(gaddr) : "memory");
}
__device__ __forceinline__ void wait_async0() {
    asm volatile("s_wait_asynccnt 0" ::: "memory");
}

// ---------- CDNA5 Tensor Data Mover: 2D tile -> LDS (TENSORcnt-tracked) ----------
// D# packing per ISA 8.3/8.4: group0 {count=1, lds_addr, global_addr, type=2},
// group1 {data_size=2B, pad_enable, pad codes, tensor dims, tile dims, stride}.
#if __has_builtin(__builtin_amdgcn_tensor_load_to_lds)
#define HAVE_TDM 1
__device__ __forceinline__ void tdm_load_2d(void* lds_ptr, const void* gaddr,
                                            unsigned tile_rows, unsigned tile_cols,
                                            unsigned tensor_cols, unsigned tensor_rows,
                                            unsigned row_stride_elems,
                                            unsigned pad_interval_code,
                                            unsigned pad_amount_code) {
    const unsigned lds = (unsigned)(unsigned long long)lds_ptr;
    const unsigned long long ga = (unsigned long long)gaddr;
    u32x4 g0;
    g0[0] = 1u;                                               // count=1 (valid user D#)
    g0[1] = lds;                                              // lds_addr
    g0[2] = (unsigned)(ga & 0xffffffffu);                     // global_addr[31:0]
    g0[3] = (unsigned)((ga >> 32) & 0x01ffffffu) | (2u << 30);// global_addr[56:32], type=2
    i32x8 g1;
    g1[0] = (int)((1u << 16) | (1u << 20) |                   // data_size=2B, pad_enable
                  (pad_interval_code << 22) | (pad_amount_code << 25));
    g1[1] = (int)((tensor_cols & 0xffffu) << 16);             // tensor_dim0[15:0] @ [63:48]
    g1[2] = (int)(((tensor_cols >> 16) & 0xffffu) |           // tensor_dim0[31:16]
                  ((tensor_rows & 0xffffu) << 16));           // tensor_dim1[15:0]
    g1[3] = (int)(((tensor_rows >> 16) & 0xffffu) |           // tensor_dim1[31:16]
                  ((tile_cols & 0xffffu) << 16));             // tile_dim0
    g1[4] = (int)(tile_rows & 0xffffu);                       // tile_dim1 (tile_dim2=0)
    g1[5] = (int)row_stride_elems;                            // tensor_dim0_stride[31:0]
    g1[6] = 0;
    g1[7] = 0;
    i32x4 z4 = {0, 0, 0, 0};
#if __clang_major__ >= 23
    i32x8 z8 = {0, 0, 0, 0, 0, 0, 0, 0};
    __builtin_amdgcn_tensor_load_to_lds(g0, g1, z4, z4, z8, 0);
#else
    __builtin_amdgcn_tensor_load_to_lds(g0, g1, z4, z4, 0);
#endif
}
#else
#define HAVE_TDM 0
#endif

// ---------- WMMA fragment helpers (layouts per CDNA5 ISA 7.12.2) ----------
// A 16x32 f16: lanes 0-15 row M=lane, K={0..7,16..23}; lanes 16-31 same rows, K={8..15,24..31}
__device__ __forceinline__ v16h frag_a(const _Float16* base, int stride, int lane) {
    const int r  = lane & 15;
    const int hi = (lane >> 4) & 1;
    const _Float16* p = base + r * stride + hi * 8;
    v8h a0 = *(const v8h*)(p);
    v8h a1 = *(const v8h*)(p + 16);
    return __builtin_shufflevector(a0, a1, 0,1,2,3,4,5,6,7,8,9,10,11,12,13,14,15);
}
// B 32x16 f16: lanes 0-15 col N=lane, K={0..7,16..23}; lanes 16-31 same cols, K={8..15,24..31}
__device__ __forceinline__ v16h frag_b(const _Float16* base, int stride, int lane) {
    const int n  = lane & 15;
    const int hi = (lane >> 4) & 1;
    v16h t;
#pragma unroll
    for (int s = 0; s < 16; s++) t[s] = base[((s & 7) + 8 * hi + 16 * (s >> 3)) * stride + n];
    return t;
}

// ---------- fp32 -> fp16 cast ----------
__global__ void cast_f32_f16(const float* __restrict__ in, _Float16* __restrict__ out, int n) {
    int i = blockIdx.x * blockDim.x + threadIdx.x;
    if (i < n) out[i] = (_Float16)in[i];
}

// ---------- LayerNorm (fp32 in, fp16 out), one row (D=1024) per 256-thread block ----------
__global__ __launch_bounds__(256) void ln_f16(const float* __restrict__ in,
                                              const float* __restrict__ gam,
                                              const float* __restrict__ bet,
                                              _Float16* __restrict__ out) {
    __shared__ float red[16];
    const int row = blockIdx.x, tid = threadIdx.x;
    const int lane = tid & 31, wave = tid >> 5;
    const float* p = in + (size_t)row * DM;
    float x[4], s = 0.f, ss = 0.f;
#pragma unroll
    for (int i = 0; i < 4; i++) { x[i] = p[tid + i * 256]; s += x[i]; ss += x[i] * x[i]; }
#pragma unroll
    for (int m = 16; m >= 1; m >>= 1) { s += __shfl_xor(s, m, 32); ss += __shfl_xor(ss, m, 32); }
    if (lane == 0) { red[wave] = s; red[8 + wave] = ss; }
    __syncthreads();
    if (tid < 32) {
        float a  = (lane < 8) ? red[lane] : 0.f;
        float b2 = (lane < 8) ? red[8 + lane] : 0.f;
#pragma unroll
        for (int m = 4; m >= 1; m >>= 1) { a += __shfl_xor(a, m, 32); b2 += __shfl_xor(b2, m, 32); }
        if (lane == 0) { red[0] = a; red[8] = b2; }
    }
    __syncthreads();
    const float mu   = red[0] * (1.0f / DM);
    const float var  = red[8] * (1.0f / DM) - mu * mu;
    const float rstd = rsqrtf(var + 1e-5f);
#pragma unroll
    for (int i = 0; i < 4; i++) {
        int c = tid + i * 256;
        out[(size_t)row * DM + c] = (_Float16)((x[i] - mu) * rstd * gam[c] + bet[c]);
    }
}

// ---------- WMMA GEMM: C[M,N] = A[M,K](f16) @ W[K,N](f16) + bias, fused epilogue ----------
// Tile staging via TDM tensor_load_to_lds (one wave issues 2 descriptors per
// K-step; TDM applies the bank-conflict padding). EPI 0: f16 store,
// EPI 1: exact GELU -> f16, EPI 2: + residual -> f32.
template <int EPI>
__global__ __launch_bounds__(256) void gemm_wmma(const _Float16* __restrict__ A,
                                                 const _Float16* __restrict__ Bw,
                                                 const float* __restrict__ bias,
                                                 const float* __restrict__ res,
                                                 _Float16* __restrict__ out_h,
                                                 float* __restrict__ out_f,
                                                 int M, int N, int K) {
    __shared__ __align__(16) _Float16 lA[128 * 40];   // 128 x 32, padded stride 40
    __shared__ __align__(16) _Float16 lB[32 * 136];   // 32 x 128, padded stride 136
    const int tid = threadIdx.x, lane = tid & 31, wave = tid >> 5;
    const int wm = wave & 3, wn = wave >> 2;          // 4x2 wave grid; wave tile 32x64
    const int m0 = blockIdx.y * 128, n0 = blockIdx.x * 128;

    v8f acc[2][4];
#pragma unroll
    for (int i = 0; i < 2; i++)
#pragma unroll
        for (int j = 0; j < 4; j++) acc[i][j] = {};

    for (int k0 = 0; k0 < K; k0 += 32) {
#if HAVE_TDM
        if (wave == 0) {
            // A tile: 128 rows x 32 elems, pad 4 DWORDs per 16 DWORDs -> stride 40
            tdm_load_2d(lA, A + (size_t)m0 * K + k0, 128, 32,
                        (unsigned)K, (unsigned)M, (unsigned)K, 3u, 3u);
            // B tile: 32 rows x 128 elems, pad 4 DWORDs per 64 DWORDs -> stride 136
            tdm_load_2d(lB, Bw + (size_t)k0 * N + n0, 32, 128,
                        (unsigned)N, (unsigned)K, (unsigned)N, 5u, 3u);
        }
        __builtin_amdgcn_s_wait_tensorcnt(0);
#else
        {   // fallback: per-thread async DMA global -> LDS
            const int arow = tid >> 1, aoff = (tid & 1) * 16;
            const int brow = tid >> 3, boff = (tid & 7) * 16;
            const _Float16* ga = A  + (size_t)(m0 + arow) * K + k0 + aoff;
            const _Float16* gb = Bw + (size_t)(k0 + brow) * N + n0 + boff;
            async_b128(lA + arow * 40 + aoff,      ga);
            async_b128(lA + arow * 40 + aoff + 8,  ga + 8);
            async_b128(lB + brow * 136 + boff,     gb);
            async_b128(lB + brow * 136 + boff + 8, gb + 8);
            wait_async0();
        }
#endif
        if (k0 + 32 < K) {                       // global_prefetch_b8 next tiles
            __builtin_prefetch(A + (size_t)(m0 + (tid >> 1)) * K + k0 + 32, 0, 1);
            __builtin_prefetch(Bw + (size_t)(k0 + 32 + (tid >> 3)) * N + n0, 0, 1);
        }
        __syncthreads();
        v16h af[2], bf[4];
#pragma unroll
        for (int i = 0; i < 2; i++) af[i] = frag_a(lA + (wm * 32 + i * 16) * 40, 40, lane);
#pragma unroll
        for (int j = 0; j < 4; j++) bf[j] = frag_b(lB + wn * 64 + j * 16, 136, lane);
#pragma unroll
        for (int i = 0; i < 2; i++)
#pragma unroll
            for (int j = 0; j < 4; j++)
                acc[i][j] = __builtin_amdgcn_wmma_f32_16x16x32_f16(
                    false, af[i], false, bf[j], (short)0, acc[i][j], false, false);
        __syncthreads();
    }

    const int hi = (lane >> 4) & 1, ln = lane & 15;
#pragma unroll
    for (int i = 0; i < 2; i++)
#pragma unroll
        for (int j = 0; j < 4; j++)
#pragma unroll
            for (int r = 0; r < 8; r++) {
                const int row = m0 + wm * 32 + i * 16 + r + 8 * hi;
                const int col = n0 + wn * 64 + j * 16 + ln;
                const size_t idx = (size_t)row * N + col;
                float x = acc[i][j][r] + bias[col];
                if (EPI == 0) {
                    out_h[idx] = (_Float16)x;
                } else if (EPI == 1) {
                    out_h[idx] = (_Float16)(0.5f * x * (1.0f + erff(x * 0.70710678118654752f)));
                } else {
                    out_f[idx] = res[idx] + x;
                }
            }
}

// ---------- flash attention with null-key; Hres = softmax(QK^T/32)V + X ----------
// S^T = K @ Q^T so per-tile WMMA operands (K, V, P) are contiguous LDS loads;
// 64 full KV tiles run mask-free with async-to-LDS staging, one peeled tail
// tile handles the null key / zero padding / masking.
__global__ __launch_bounds__(128) void attn_wmma(const _Float16* __restrict__ Q,
                                                 const _Float16* __restrict__ Kh,
                                                 const _Float16* __restrict__ Vh,
                                                 const _Float16* __restrict__ nk,
                                                 const float* __restrict__ X,
                                                 float* __restrict__ Hres) {
    __shared__ __align__(16) _Float16 lK[32 * 72];      // 32 kv x 64 feat, padded
    __shared__ __align__(16) _Float16 lV[32 * 72];
    __shared__ __align__(16) _Float16 lP[4 * 16 * 40];  // per-wave P [q][kv], padded stride 40
    const int tid = threadIdx.x, lane = tid & 31, wv = tid >> 5;
    const int nqb  = NQS / 64;
    const int b    = blockIdx.x / (NH * nqb);
    const int h    = (blockIdx.x / nqb) % NH;
    const int qblk = blockIdx.x % nqb;
    const int qb   = qblk * 64 + wv * 16;
    const int hi = (lane >> 4) & 1, ln = lane & 15;

    // Q^T B-fragments: lane%16 = q column, elements = feature k. Built once, reused 65x.
    v16h bq[2];
#pragma unroll
    for (int kc = 0; kc < 2; kc++) {
        v16h t;
#pragma unroll
        for (int n = 0; n < 16; n++)
            t[n] = Q[(size_t)(b * NQS + qb + ln) * DM + h * DH + kc * 32 + (n & 7) + 8 * hi + 16 * (n >> 3)];
        bq[kc] = t;
    }

    v8f o[4];
#pragma unroll
    for (int f = 0; f < 4; f++) o[f] = {};
    float m = -3.0e38f, l = 0.f;     // per-lane softmax state for q column (qb + ln)

    const int crow = tid >> 2, coff = (tid & 3) * 16;
    _Float16* myP = lP + wv * 16 * 40;

    // one KV tile: S^T -> online softmax -> O += P@V   (mask only in tail tile)
    auto tile_compute = [&](int kv0, bool mask) {
        v8f s[2]; s[0] = {}; s[1] = {};
#pragma unroll
        for (int ns = 0; ns < 2; ns++)
#pragma unroll
            for (int kc = 0; kc < 2; kc++) {
                v16h ak = frag_a(lK + (ns * 16) * 72 + kc * 32, 72, lane);
                s[ns] = __builtin_amdgcn_wmma_f32_16x16x32_f16(
                    false, ak, false, bq[kc], (short)0, s[ns], false, false);
            }
#pragma unroll
        for (int ns = 0; ns < 2; ns++)
#pragma unroll
            for (int r = 0; r < 8; r++) {
                const float v = s[ns][r] * 0.03125f;      // 1/sqrt(D)
                if (mask) {
                    const int kvi = kv0 + ns * 16 + r + 8 * hi;
                    s[ns][r] = (kvi < NKV1) ? v : -3.0e38f;
                } else {
                    s[ns][r] = v;
                }
            }
        float x = -3.0e38f;
#pragma unroll
        for (int ns = 0; ns < 2; ns++)
#pragma unroll
            for (int r = 0; r < 8; r++) x = fmaxf(x, s[ns][r]);
        x = fmaxf(x, __shfl_xor(x, 16, 32));
        const float mn = fmaxf(m, x);
        const float corr = __expf(m - mn);
        float rs = 0.f;
#pragma unroll
        for (int ns = 0; ns < 2; ns++)
#pragma unroll
            for (int r = 0; r < 8; r++) {
                const float p = __expf(s[ns][r] - mn);
                rs += p;
                myP[ln * 40 + ns * 16 + r + 8 * hi] = (_Float16)p;   // P[q][kv]
            }
        rs += __shfl_xor(rs, 16, 32);
        l = l * corr + rs;
        m = mn;
        float corrb[8];
#pragma unroll
        for (int r = 0; r < 8; r++) corrb[r] = __shfl(corr, r + 8 * hi, 32);
#pragma unroll
        for (int f = 0; f < 4; f++)
#pragma unroll
            for (int r = 0; r < 8; r++) o[f][r] *= corrb[r];
        v16h pf = frag_a(myP, 40, lane);
#pragma unroll
        for (int f = 0; f < 4; f++) {
            v16h vf = frag_b(lV + f * 16, 72, lane);
            o[f] = __builtin_amdgcn_wmma_f32_16x16x32_f16(
                false, pf, false, vf, (short)0, o[f], false, false);
        }
    };

    // ---- 64 full tiles: async DMA staging, no guards, no masking ----
    for (int kv0 = 0; kv0 < NKVS; kv0 += 32) {
        const _Float16* kp = Kh + (size_t)(b * NKVS + kv0 + crow) * DM + h * DH + coff;
        const _Float16* vp = Vh + (size_t)(b * NKVS + kv0 + crow) * DM + h * DH + coff;
        async_b128(lK + crow * 72 + coff,     kp);
        async_b128(lK + crow * 72 + coff + 8, kp + 8);
        async_b128(lV + crow * 72 + coff,     vp);
        async_b128(lV + crow * 72 + coff + 8, vp + 8);
        wait_async0();
        __syncthreads();
        tile_compute(kv0, false);
        __syncthreads();
    }

    // ---- tail tile (kv0 = NKVS): row 2048 = null key / zero value, rest masked ----
    {
        v8h k0v = {}, k1v = {}, v0v = {}, v1v = {};
        if (crow == 0) {               // kv index NKVS -> learned null key, value 0
            const _Float16* kp = nk + h * DH + coff;
            k0v = *(const v8h*)kp; k1v = *(const v8h*)(kp + 8);
        }
        *(v8h*)(lK + crow * 72 + coff)     = k0v;
        *(v8h*)(lK + crow * 72 + coff + 8) = k1v;
        *(v8h*)(lV + crow * 72 + coff)     = v0v;
        *(v8h*)(lV + crow * 72 + coff + 8) = v1v;
        __syncthreads();
        tile_compute(NKVS, true);
    }

    // Hres = O/l + X  (l broadcast from owning lane per row)
    float lb[8];
#pragma unroll
    for (int r = 0; r < 8; r++) lb[r] = __shfl(l, r + 8 * hi, 32);
#pragma unroll
    for (int f = 0; f < 4; f++)
#pragma unroll
        for (int r = 0; r < 8; r++) {
            const int q   = qb + r + 8 * hi;
            const int col = h * DH + f * 16 + ln;
            const size_t idx = (size_t)(b * NQS + q) * DM + col;
            Hres[idx] = o[f][r] / lb[r] + X[idx];
        }
}

extern "C" void kernel_launch(void* const* d_in, const int* in_sizes, int n_in,
                              void* d_out, int out_size, void* d_ws, size_t ws_size,
                              hipStream_t stream) {
    const float* X    = (const float*)d_in[0];
    const float* Y    = (const float*)d_in[1];
    const float* Wq   = (const float*)d_in[2];
    const float* bq   = (const float*)d_in[3];
    const float* Wk   = (const float*)d_in[4];
    const float* bk   = (const float*)d_in[5];
    const float* Wv   = (const float*)d_in[6];
    const float* bv   = (const float*)d_in[7];
    const float* nlk  = (const float*)d_in[8];
    const float* g0   = (const float*)d_in[9];
    const float* b0   = (const float*)d_in[10];
    const float* g0kv = (const float*)d_in[11];
    const float* b0kv = (const float*)d_in[12];
    const float* g1   = (const float*)d_in[13];
    const float* b1   = (const float*)d_in[14];
    const float* W1   = (const float*)d_in[15];
    const float* bf1  = (const float*)d_in[16];
    const float* W2   = (const float*)d_in[17];
    const float* bf2  = (const float*)d_in[18];
    float* out = (float*)d_out;

    char* ws = (char*)d_ws;
    size_t off = 0;
    auto alloc = [&](size_t bytes) {
        char* p = ws + off;
        off = (off + bytes + 255) & ~(size_t)255;
        return p;
    };
    const size_t M = (size_t)NB * NQS;                // 4096 rows
    _Float16* Wq_h = (_Float16*)alloc((size_t)DM * DM * 2);
    _Float16* Wk_h = (_Float16*)alloc((size_t)DM * DM * 2);
    _Float16* Wv_h = (_Float16*)alloc((size_t)DM * DM * 2);
    _Float16* W1_h = (_Float16*)alloc((size_t)DM * 4 * DM * 2);
    _Float16* W2_h = (_Float16*)alloc((size_t)4 * DM * DM * 2);
    _Float16* nk_h = (_Float16*)alloc((size_t)DM * 2);
    _Float16* Xn_h = (_Float16*)alloc(M * DM * 2);
    _Float16* Yn_h = (_Float16*)alloc(M * DM * 2);
    _Float16* Qh   = (_Float16*)alloc(M * DM * 2);
    _Float16* Kh   = (_Float16*)alloc(M * DM * 2);
    _Float16* Vh   = (_Float16*)alloc(M * DM * 2);
    float*    Hres = (float*)   alloc(M * DM * 4);
    _Float16* Hr_h = (_Float16*)alloc(M * DM * 2);
    _Float16* F1_h = (_Float16*)alloc(M * 4 * DM * 2);

    const int DD = DM * DM;
    cast_f32_f16<<<(DD + 255) / 256, 256, 0, stream>>>(Wq, Wq_h, DD);
    cast_f32_f16<<<(DD + 255) / 256, 256, 0, stream>>>(Wk, Wk_h, DD);
    cast_f32_f16<<<(DD + 255) / 256, 256, 0, stream>>>(Wv, Wv_h, DD);
    cast_f32_f16<<<(4 * DD + 255) / 256, 256, 0, stream>>>(W1, W1_h, 4 * DD);
    cast_f32_f16<<<(4 * DD + 255) / 256, 256, 0, stream>>>(W2, W2_h, 4 * DD);
    cast_f32_f16<<<(DM + 255) / 256, 256, 0, stream>>>(nlk, nk_h, DM);

    ln_f16<<<(int)M, 256, 0, stream>>>(X, g0, b0, Xn_h);
    ln_f16<<<(int)M, 256, 0, stream>>>(Y, g0kv, b0kv, Yn_h);

    dim3 gQ(DM / 128, (int)M / 128);                  // (8, 32)
    gemm_wmma<0><<<gQ, 256, 0, stream>>>(Xn_h, Wq_h, bq, nullptr, Qh, nullptr, (int)M, DM, DM);
    gemm_wmma<0><<<gQ, 256, 0, stream>>>(Yn_h, Wk_h, bk, nullptr, Kh, nullptr, (int)M, DM, DM);
    gemm_wmma<0><<<gQ, 256, 0, stream>>>(Yn_h, Wv_h, bv, nullptr, Vh, nullptr, (int)M, DM, DM);

    attn_wmma<<<NB * NH * (NQS / 64), 128, 0, stream>>>(Qh, Kh, Vh, nk_h, X, Hres);

    ln_f16<<<(int)M, 256, 0, stream>>>(Hres, g1, b1, Hr_h);

    dim3 gF1(4 * DM / 128, (int)M / 128);             // (32, 32)
    gemm_wmma<1><<<gF1, 256, 0, stream>>>(Hr_h, W1_h, bf1, nullptr, F1_h, nullptr, (int)M, 4 * DM, DM);
    dim3 gF2(DM / 128, (int)M / 128);                 // (8, 32)
    gemm_wmma<2><<<gF2, 256, 0, stream>>>(F1_h, W2_h, bf2, Hres, nullptr, out, (int)M, DM, 4 * DM);
}
